// MHAttention_15771119911171
// MI455X (gfx1250) — compile-verified, matrix operand order checked by reading, not run on hardware
//
#include <hip/hip_runtime.h>
#include <stdint.h>

// Problem constants (match reference)
#define BB 8
#define SS 1024
#define DD 1024
#define HH 16
#define DHD 64

typedef __attribute__((ext_vector_type(16))) _Float16 v16h;
typedef __attribute__((ext_vector_type(8)))  _Float16 v8h;
typedef __attribute__((ext_vector_type(8)))  float    v8f;
typedef __attribute__((ext_vector_type(4)))  unsigned int v4u;
typedef __attribute__((ext_vector_type(8)))  int      v8i;
typedef __attribute__((ext_vector_type(4)))  int      v4i;

// WMMA D = A*B + C, f16 inputs, f32 accumulate (codegen-confirmed builtin)
__device__ __forceinline__ v8f wmma_f32_f16(v16h a, v16h b, v8f c) {
  return __builtin_amdgcn_wmma_f32_16x16x32_f16(false, a, false, b, (short)0, c,
                                                false, false);
}

// Load a 16-bit A/B fragment (16x32, K-major per ISA layout) for this lane.
// Lane L holds row (caller passes row incl. lane%16); halves K = {8*hi..} and
// {16+8*hi..} -> two contiguous 16-byte loads. Works for global or LDS ptrs.
__device__ __forceinline__ v16h frag_load(const _Float16* base, int row,
                                          int stride_h, int k0) {
  const int hi8 = (threadIdx.x & 16) >> 1;  // 0 or 8
  const _Float16* p = base + (size_t)row * stride_h + k0 + hi8;
  v8h lo = *(const v8h*)(p);
  v8h hh = *(const v8h*)(p + 16);
  v16h f;
#pragma unroll
  for (int i = 0; i < 8; ++i) { f[i] = lo[i]; f[i + 8] = hh[i]; }
  return f;
}

// ---------------------------------------------------------------------------
// Tensor Data Mover staging (CDNA5): async DMA of a 64x64 f16 2D tile from
// row-strided global memory into LDS, with hardware padding producing a
// 72-half (144B) LDS row stride (pad 4 DWORDs every 32 DWORDs).
// ---------------------------------------------------------------------------
#if __has_builtin(__builtin_amdgcn_tensor_load_to_lds)
#define USE_TDM 1

__device__ __forceinline__ uint32_t lds_byte_addr(const _Float16* p) {
  return (uint32_t)(uintptr_t)(__attribute__((address_space(3))) const _Float16*)p;
}

__device__ __forceinline__ void tdm_load_tile64(const _Float16* gptr,
                                                const _Float16* lptr,
                                                uint32_t stride0_elems) {
  const uint64_t ga = (uint64_t)(uintptr_t)gptr;
  v4u g0;
  g0.x = 1u;                                   // count=1, user descriptor
  g0.y = lds_byte_addr(lptr);                  // lds_addr (bytes)
  g0.z = (uint32_t)ga;                         // global_addr[31:0]
  g0.w = ((uint32_t)(ga >> 32) & 0x01ffffffu)  // global_addr[56:32]
         | (2u << 30);                         // type=2 ("image")
  v8i g1;
  // data_size=2B | pad_enable | pad_interval=4 (32 DW) | pad_amount=3 (4 DW)
  g1[0] = (int)((1u << 16) | (1u << 20) | (4u << 22) | (3u << 25));
  g1[1] = (int)(64u << 16);          // tensor_dim0[15:0] (bits 63:48)
  g1[2] = (int)(64u << 16);          // tensor_dim1[15:0] (bits 95:80)
  g1[3] = (int)(64u << 16);          // tile_dim0 (bits 127:112)
  g1[4] = 64;                        // tile_dim1 (tile_dim2=0 -> 2D)
  g1[5] = (int)stride0_elems;        // tensor_dim0_stride[31:0]
  g1[6] = 0;
  g1[7] = 0;
  v4i gz = {0, 0, 0, 0};
#if __has_include(<hip/amd_detail/amd_gfx1250_TDM.h>)
  v8i gz8 = {0, 0, 0, 0, 0, 0, 0, 0};
  __builtin_amdgcn_tensor_load_to_lds(g0, g1, gz, gz, gz8, 0);
#else
  __builtin_amdgcn_tensor_load_to_lds(g0, g1, gz, gz, 0);
#endif
}
#else
#define USE_TDM 0
#endif

// ---------------------------------------------------------------------------
// Kernel 1: fused projection GEMMs.  C = X @ W   (M=B*S=8192, N=K=1024)
// blockIdx.z: 0 -> Q (f16 row-major), 1 -> K (f16 row-major),
//             2 -> V (f16 transposed per head: [B,H,DH,S])
// Block tile 64(M) x 128(N), 8 waves each computing 32x32 (4 WMMA / K-step).
// ---------------------------------------------------------------------------
__global__ __launch_bounds__(256) void proj_kernel(
    const float* __restrict__ Xq, const float* __restrict__ Xk,
    const float* __restrict__ Xv, const float* __restrict__ Wq,
    const float* __restrict__ Wk, const float* __restrict__ Wv,
    _Float16* __restrict__ Qh, _Float16* __restrict__ Kh,
    _Float16* __restrict__ Vt) {
  constexpr int AST = 40;  // 80B row stride: 16B aligned, conflict-free
  constexpr int BST = 40;
  __shared__ _Float16 sa[64 * AST];    // A tile: 64 rows x 32 k (row-major)
  __shared__ _Float16 sb[128 * BST];   // B tile transposed: sb[n][k]

  const int mode = blockIdx.z;
  const float* X = (mode == 0) ? Xq : (mode == 1) ? Xk : Xv;
  const float* W = (mode == 0) ? Wq : (mode == 1) ? Wk : Wv;

  const int m0 = blockIdx.y * 64;
  const int n0 = blockIdx.x * 128;
  const int tid = threadIdx.x;
  const int lane = tid & 31, wid = tid >> 5;
  const int wy = wid >> 2, wx = wid & 3;   // wave grid 2(M) x 4(N)
  const int l15 = lane & 15;
  const int hi = lane >> 4;

  v8f acc[2][2] = {};

  for (int k0 = 0; k0 < DD; k0 += 32) {
    __syncthreads();  // protect LDS tiles from previous iteration's readers
    // stage A tile (64x32 f32 -> f16)
#pragma unroll
    for (int it = 0; it < 2; ++it) {
      int idx = tid + it * 256;                 // 512 float4 chunks
      int row = idx >> 3, c4 = (idx & 7) * 4;
      float4 xv = *(const float4*)(X + (size_t)(m0 + row) * DD + k0 + c4);
      _Float16* d = sa + row * AST + c4;
      d[0] = (_Float16)xv.x; d[1] = (_Float16)xv.y;
      d[2] = (_Float16)xv.z; d[3] = (_Float16)xv.w;
    }
    // stage B tile (32x128 f32), store transposed: sb[n][k]
#pragma unroll
    for (int it = 0; it < 4; ++it) {
      int idx = tid + it * 256;                 // 1024 float4 chunks
      int row = idx >> 5, c4 = (idx & 31) * 4;
      float4 wv = *(const float4*)(W + (size_t)(k0 + row) * DD + n0 + c4);
      sb[(c4 + 0) * BST + row] = (_Float16)wv.x;
      sb[(c4 + 1) * BST + row] = (_Float16)wv.y;
      sb[(c4 + 2) * BST + row] = (_Float16)wv.z;
      sb[(c4 + 3) * BST + row] = (_Float16)wv.w;
    }
    __syncthreads();

    v16h a0 = frag_load(sa, wy * 32 + l15,      AST, 0);
    v16h a1 = frag_load(sa, wy * 32 + 16 + l15, AST, 0);
    v16h b0 = frag_load(sb, wx * 32 + l15,      BST, 0);
    v16h b1 = frag_load(sb, wx * 32 + 16 + l15, BST, 0);
    acc[0][0] = wmma_f32_f16(a0, b0, acc[0][0]);
    acc[0][1] = wmma_f32_f16(a0, b1, acc[0][1]);
    acc[1][0] = wmma_f32_f16(a1, b0, acc[1][0]);
    acc[1][1] = wmma_f32_f16(a1, b1, acc[1][1]);
  }

  // C layout: vgpr r -> row r + 8*(lane/16), col = lane%16
#pragma unroll
  for (int i = 0; i < 2; ++i)
#pragma unroll
    for (int j = 0; j < 2; ++j)
#pragma unroll
      for (int r = 0; r < 8; ++r) {
        int rowg = m0 + wy * 32 + i * 16 + r + 8 * hi;
        int colg = n0 + wx * 32 + j * 16 + l15;
        _Float16 val = (_Float16)acc[i][j][r];
        if (mode < 2) {
          _Float16* dst = (mode == 0) ? Qh : Kh;
          dst[(size_t)rowg * DD + colg] = val;
        } else {  // V transposed per head: [B,H,DH,S]
          int b = rowg >> 10, s = rowg & 1023;
          int hh = colg >> 6, dh = colg & 63;
          Vt[(((size_t)b * HH + hh) * DHD + dh) * SS + s] = val;
        }
      }
}

// ---------------------------------------------------------------------------
// Kernel 2: flash attention with per-query valid_len mask (-1000 fill).
// Block = 4 waves = 64 q-rows for one (b,h); key loop in 64-wide tiles,
// double-buffered in LDS via the Tensor Data Mover (TENSORcnt pipelining).
// ---------------------------------------------------------------------------
__global__ __launch_bounds__(128) void attn_kernel(
    const _Float16* __restrict__ Qh, const _Float16* __restrict__ Kh,
    const _Float16* __restrict__ Vt, const int* __restrict__ vlen,
    float* __restrict__ out) {
  constexpr int KST = 72;     // 144B row stride: 16B aligned, conflict-free
  constexpr int TILE = 64 * KST;
  // [ K buf0 | K buf1 | V buf0 | V buf1 | P (4 waves x 16 rows) ]
  __shared__ _Float16 smem[5 * TILE];

  const int QT = SS / 64;
  const int blk = blockIdx.x;
  const int b = blk / (HH * QT);
  const int h = (blk / QT) % HH;
  const int qbase = (blk % QT) * 64;

  const int tid = threadIdx.x;
  const int lane = tid & 31, wid = tid >> 5;
  const int l15 = lane & 15;
  const int hi = lane >> 4;
  const int qrow0 = qbase + wid * 16;

  // Q fragments live in registers for the whole kernel (K-dim = 64 = 2 frags)
  const _Float16* qb = Qh + (size_t)(b * SS + qrow0) * DD + h * DHD;
  v16h qf0 = frag_load(qb, l15, DD, 0);
  v16h qf1 = frag_load(qb, l15, DD, 32);

  int vl[8];
#pragma unroll
  for (int r = 0; r < 8; ++r) vl[r] = vlen[b * SS + qrow0 + r + 8 * hi];

  v8f acc[4] = {};
  float m[8], l[8];
#pragma unroll
  for (int r = 0; r < 8; ++r) { m[r] = -1.0e30f; l[r] = 0.0f; }

  const float scale = 0.125f;  // 1/sqrt(DH)
  const _Float16* Kb = Kh + (size_t)b * SS * DD + h * DHD;
  const _Float16* Vb = Vt + ((size_t)b * HH + h) * DHD * SS;

#if USE_TDM
  if (wid == 0) {  // wave-level DMA: prologue loads for tile 0
    tdm_load_tile64(Kb, smem, DD);
    tdm_load_tile64(Vb, smem + 2 * TILE, SS);
  }
#endif

  for (int t = 0; t < SS / 64; ++t) {
    const int kbase = t * 64;
    _Float16* sk = smem + (t & 1) * TILE;             // K buffer for tile t
    _Float16* sv = smem + (2 + (t & 1)) * TILE;       // V buffer for tile t

#if USE_TDM
    if (wid == 0) {
      if (t + 1 < SS / 64) {  // issue next tile into the other buffer
        tdm_load_tile64(Kb + (size_t)(kbase + 64) * DD,
                        smem + ((t + 1) & 1) * TILE, DD);
        tdm_load_tile64(Vb + (kbase + 64),
                        smem + (2 + ((t + 1) & 1)) * TILE, SS);
        // in-order TENSORcnt: <=2 outstanding => tile t's two loads are done
        __builtin_amdgcn_s_wait_tensorcnt(2);
      } else {
        __builtin_amdgcn_s_wait_tensorcnt(0);  // last tile: drain everything
      }
    }
    __syncthreads();  // publish tile t to all waves
#else
    // fallback staging: global_load_b128 -> ds_store_b128
#pragma unroll
    for (int it = 0; it < 4; ++it) {
      int idx = tid + it * 128;                 // 512 chunks of 8 halves
      int row = idx >> 3, c8 = (idx & 7) * 8;
      *(v8h*)(sk + row * KST + c8) =
          *(const v8h*)(Kb + (size_t)(kbase + row) * DD + c8);
      *(v8h*)(sv + row * KST + c8) =
          *(const v8h*)(Vb + (size_t)row * SS + kbase + c8);
    }
    if (kbase + 64 < SS) {
      __builtin_prefetch(Kb + (size_t)(kbase + 64 + (tid >> 1)) * DD +
                             (tid & 1) * 32, 0, 1);
      __builtin_prefetch(Vb + (size_t)(tid >> 1) * SS + kbase + 64 +
                             (tid & 1) * 32, 0, 1);
    }
    __syncthreads();
#endif

    // scores S = scale * Q K^T, masked with -1000 where key >= valid_len[q]
    float sc[4][8];
#pragma unroll
    for (int n = 0; n < 4; ++n) {
      v8f c = {};
      c = wmma_f32_f16(qf0, frag_load(sk, n * 16 + l15, KST, 0), c);
      c = wmma_f32_f16(qf1, frag_load(sk, n * 16 + l15, KST, 32), c);
      int kcol = kbase + n * 16 + l15;
#pragma unroll
      for (int r = 0; r < 8; ++r) {
        float sval = c[r] * scale;
        sc[n][r] = (kcol < vl[r]) ? sval : -1000.0f;
      }
    }

    // online softmax (rows live in 16-lane halves -> xor-shuffle reductions)
    _Float16* spw = smem + 4 * TILE + wid * 16 * KST;
#pragma unroll
    for (int r = 0; r < 8; ++r) {
      float mx = fmaxf(fmaxf(sc[0][r], sc[1][r]), fmaxf(sc[2][r], sc[3][r]));
#pragma unroll
      for (int o = 1; o < 16; o <<= 1) mx = fmaxf(mx, __shfl_xor(mx, o, 32));
      float mnew = fmaxf(m[r], mx);
      float sf = __expf(m[r] - mnew);
      float rs = 0.0f;
#pragma unroll
      for (int n = 0; n < 4; ++n) {
        float p = __expf(sc[n][r] - mnew);
        sc[n][r] = p;
        rs += p;
      }
#pragma unroll
      for (int o = 1; o < 16; o <<= 1) rs += __shfl_xor(rs, o, 32);
      l[r] = l[r] * sf + rs;
      m[r] = mnew;
#pragma unroll
      for (int d = 0; d < 4; ++d) acc[d][r] *= sf;
    }

    // round-trip P through wave-private LDS to re-fragment as WMMA A-matrix
#pragma unroll
    for (int n = 0; n < 4; ++n)
#pragma unroll
      for (int r = 0; r < 8; ++r)
        spw[(r + 8 * hi) * KST + n * 16 + l15] = (_Float16)sc[n][r];
    __asm__ volatile("s_wait_dscnt 0" ::: "memory");

    v16h pf0 = frag_load(spw, l15, KST, 0);
    v16h pf1 = frag_load(spw, l15, KST, 32);
#pragma unroll
    for (int d = 0; d < 4; ++d) {
      acc[d] = wmma_f32_f16(pf0, frag_load(sv, d * 16 + l15, KST, 0), acc[d]);
      acc[d] = wmma_f32_f16(pf1, frag_load(sv, d * 16 + l15, KST, 32), acc[d]);
    }
    __syncthreads();  // compute done: next iteration may overwrite buffers
  }

  // O = acc / l, write f32 [B,S,D]
  float* ob = out + (size_t)(b * SS + qrow0) * DD + h * DHD;
#pragma unroll
  for (int d = 0; d < 4; ++d)
#pragma unroll
    for (int r = 0; r < 8; ++r) {
      float inv = 1.0f / l[r];
      ob[(size_t)(r + 8 * hi) * DD + d * 16 + l15] = acc[d][r] * inv;
    }
}

extern "C" void kernel_launch(void* const* d_in, const int* in_sizes, int n_in,
                              void* d_out, int out_size, void* d_ws,
                              size_t ws_size, hipStream_t stream) {
  (void)in_sizes; (void)n_in; (void)out_size; (void)ws_size;
  const float* q  = (const float*)d_in[0];
  const float* k  = (const float*)d_in[1];
  const float* v  = (const float*)d_in[2];
  const int* vl   = (const int*)d_in[3];
  const float* Wq = (const float*)d_in[4];
  const float* Wk = (const float*)d_in[5];
  const float* Wv = (const float*)d_in[6];

  // workspace: Q f16 (16MB) | K f16 (16MB) | V^T f16 (16MB)
  _Float16* ws = (_Float16*)d_ws;
  _Float16* Qh = ws;
  _Float16* Kh = ws + (size_t)BB * SS * DD;
  _Float16* Vt = ws + (size_t)2 * BB * SS * DD;

  dim3 gp(DD / 128, (BB * SS) / 64, 3);
  proj_kernel<<<gp, 256, 0, stream>>>(q, k, v, Wq, Wk, Wv, Qh, Kh, Vt);

  dim3 ga(BB * HH * (SS / 64));
  attn_kernel<<<ga, 128, 0, stream>>>(Qh, Kh, Vt, vl, (float*)d_out);
}
